// LinearMoE_47244640256352
// MI455X (gfx1250) — matrix operand order repaired and checked by reading, not compiled, not run on hardware
//
#include <hip/hip_runtime.h>

// ---------------- problem constants ----------------
#define N_TOK   8192
#define D_IN    2048
#define HID     2048
#define N_EXP   8

// GEMM tiling
#define BM 128
#define BN 128
#define BK 32
#define KT (D_IN / BK)   // 64 K-steps
#define LDA 40   // padded bf16 row stride (80B: 16B-aligned, conflict-free b128 reads)

typedef __attribute__((ext_vector_type(16))) __bf16 v16bf;
typedef __attribute__((ext_vector_type(8)))  __bf16 v8bf;
typedef __attribute__((ext_vector_type(4)))  __bf16 v4bf;
typedef __attribute__((ext_vector_type(8)))  float  v8f;

// ---------------------------------------------------------------------------
// Kernel 1: zero output + routing counters
// ---------------------------------------------------------------------------
__global__ __launch_bounds__(256) void moe_zero_kernel(float* __restrict__ y,
                                                       int* __restrict__ counts) {
    size_t i = ((size_t)blockIdx.x * 256 + threadIdx.x) * 4;
    if (i < (size_t)N_TOK * HID) {
        float4 z = make_float4(0.f, 0.f, 0.f, 0.f);
        *(float4*)(y + i) = z;
    }
    if (blockIdx.x == 0 && threadIdx.x < N_EXP) counts[threadIdx.x] = 0;
}

// ---------------------------------------------------------------------------
// Kernel 2: gating (softmax over 8 experts, top-2, compact per-expert lists)
// One wave (32 lanes) per token. Wg is only 64KB -> L2 resident.
// ---------------------------------------------------------------------------
__global__ __launch_bounds__(256) void moe_gating_kernel(
    const float* __restrict__ x, const float* __restrict__ Wg,
    const float* __restrict__ bg,
    int* __restrict__ counts, int* __restrict__ tok, float* __restrict__ wgt) {

    const int tid  = threadIdx.x;
    const int lane = tid & 31;
    const int wv   = tid >> 5;
    const int t    = blockIdx.x * 8 + wv;

    const float* xr = x + (size_t)t * D_IN;

    float acc[8];
#pragma unroll
    for (int e = 0; e < 8; ++e) acc[e] = 0.f;

    for (int i = 0; i < D_IN / 32; ++i) {
        int d = i * 32 + lane;
        float xv = xr[d];
        const float* wr = Wg + (size_t)d * N_EXP;
        float4 a = *(const float4*)wr;
        float4 c = *(const float4*)(wr + 4);
        acc[0] += xv * a.x; acc[1] += xv * a.y; acc[2] += xv * a.z; acc[3] += xv * a.w;
        acc[4] += xv * c.x; acc[5] += xv * c.y; acc[6] += xv * c.z; acc[7] += xv * c.w;
    }
    // butterfly reduce across the wave (wave32)
#pragma unroll
    for (int off = 16; off >= 1; off >>= 1)
#pragma unroll
        for (int e = 0; e < 8; ++e) acc[e] += __shfl_xor(acc[e], off, 32);

    if (lane == 0) {
        float l[8], p[8];
        float mx = -3.4e38f;
#pragma unroll
        for (int e = 0; e < 8; ++e) { l[e] = acc[e] + bg[e]; mx = fmaxf(mx, l[e]); }
        float s = 0.f;
#pragma unroll
        for (int e = 0; e < 8; ++e) { p[e] = expf(l[e] - mx); s += p[e]; }
        int i0 = 0;
#pragma unroll
        for (int e = 1; e < 8; ++e) if (p[e] > p[i0]) i0 = e;
        int i1 = (i0 == 0) ? 1 : 0;
#pragma unroll
        for (int e = 0; e < 8; ++e) if (e != i0 && p[e] > p[i1]) i1 = e;
        float inv = 1.f / s;

        int pos0 = atomicAdd(&counts[i0], 1);
        tok[i0 * N_TOK + pos0] = t;
        wgt[i0 * N_TOK + pos0] = p[i0] * inv;
        int pos1 = atomicAdd(&counts[i1], 1);
        tok[i1 * N_TOK + pos1] = t;
        wgt[i1 * N_TOK + pos1] = p[i1] * inv;
    }
}

// ---------------------------------------------------------------------------
// Kernel 3: routed expert GEMM with bf16 WMMA (fp32 accumulate).
// Block = 256 threads = 8 waves; tile 128x128, K-step 32; wave owns 32x64.
// Ping-pong LDS double buffering: global loads for step k+1 issue before the
// WMMAs of step k; cvt+LDS stores land in the other buffer; ONE barrier/step.
// B staging: 4x global_load_b128 per thread (4k x 4n sub-tile), register
// transpose, 4x b64 LDS stores (bank-conflict-free with LDA=40).
// ---------------------------------------------------------------------------
__global__ __launch_bounds__(256) void moe_gemm_kernel(
    const float* __restrict__ x, const float* __restrict__ W,
    const float* __restrict__ bias,
    const int* __restrict__ counts, const int* __restrict__ tok,
    const float* __restrict__ wgt, float* __restrict__ y) {

    const int e   = blockIdx.z;
    const int cnt = counts[e];
    const int mt  = blockIdx.y;
    if (mt * BM >= cnt) return;                    // block-uniform early exit
    const int n0  = blockIdx.x * BN;

    __shared__ __bf16 As[2][BM * LDA];
    __shared__ __bf16 Bs[2][BN * LDA];
    __shared__ int    tokL[BM];
    __shared__ float  wgtL[BM];

    const int tid  = threadIdx.x;
    const int lane = tid & 31;
    const int wv   = tid >> 5;

    if (tid < BM) {
        int r = mt * BM + tid;
        bool v = (r < cnt);
        tokL[tid] = v ? tok[e * N_TOK + r] : 0;
        wgtL[tid] = v ? wgt[e * N_TOK + r] : 0.f;
    }
    __syncthreads();

    // ---- staging index precompute ----
    // A: thread covers 4 rows (rA, rA+32, ...) x 4 contiguous cols (float4 load)
    const int rA = tid >> 3;          // 0..31
    const int cA = (tid & 7) * 4;     // 0..28
    const float* arow[4];
#pragma unroll
    for (int p = 0; p < 4; ++p)
        arow[p] = x + (size_t)tokL[rA + 32 * p] * D_IN + cA;

    // B: thread covers 4 n-rows x 4 k-cols; float4 row loads + reg transpose
    const int nq = (tid >> 3) * 4;    // 0..124 (n base)
    const int kh = (tid & 7) * 4;     // 0..28  (k base)
    const float* bbase = W + (size_t)e * D_IN * HID + n0 + nq;

    v8f acc[2][4] = {};

    const int wm = (wv & 3) * 32;     // wave m-offset within block tile
    const int wn = (wv >> 2) * 64;    // wave n-offset
    const int amRow = wm + (lane & 15);
    const int khA   = (lane >> 4) * 8;    // A: lanes 16-31 hold K 8..15 / 24..31
    const int bnRow = wn + (lane & 15);
    const int khB   = (lane >> 4) * 8;    // B: lanes 16-31 hold K 8..15 / 24..31

    // staging register buffers
    float4 aReg[4];
    float4 bReg[4];

    // ---- prologue: load + stage K-step 0 into buffer 0 ----
#pragma unroll
    for (int p = 0; p < 4; ++p) aReg[p] = *(const float4*)(arow[p] + 0);
#pragma unroll
    for (int i = 0; i < 4; ++i)
        bReg[i] = *(const float4*)(bbase + (size_t)(kh + i) * HID);
#pragma unroll
    for (int p = 0; p < 4; ++p) {
        v4bf h = { (__bf16)aReg[p].x, (__bf16)aReg[p].y, (__bf16)aReg[p].z, (__bf16)aReg[p].w };
        *(v4bf*)&As[0][(rA + 32 * p) * LDA + cA] = h;
    }
    {
        v4bf h0 = { (__bf16)bReg[0].x, (__bf16)bReg[1].x, (__bf16)bReg[2].x, (__bf16)bReg[3].x };
        v4bf h1 = { (__bf16)bReg[0].y, (__bf16)bReg[1].y, (__bf16)bReg[2].y, (__bf16)bReg[3].y };
        v4bf h2 = { (__bf16)bReg[0].z, (__bf16)bReg[1].z, (__bf16)bReg[2].z, (__bf16)bReg[3].z };
        v4bf h3 = { (__bf16)bReg[0].w, (__bf16)bReg[1].w, (__bf16)bReg[2].w, (__bf16)bReg[3].w };
        *(v4bf*)&Bs[0][(nq + 0) * LDA + kh] = h0;
        *(v4bf*)&Bs[0][(nq + 1) * LDA + kh] = h1;
        *(v4bf*)&Bs[0][(nq + 2) * LDA + kh] = h2;
        *(v4bf*)&Bs[0][(nq + 3) * LDA + kh] = h3;
    }
    __syncthreads();

    for (int kt = 0; kt < KT; ++kt) {
        const int cur = kt & 1;
        const int nxt = cur ^ 1;

        // ---- prefetch next K-slice from global into registers ----
        if (kt + 1 < KT) {
            const int k0 = (kt + 1) * BK;
#pragma unroll
            for (int p = 0; p < 4; ++p) aReg[p] = *(const float4*)(arow[p] + k0);
#pragma unroll
            for (int i = 0; i < 4; ++i)
                bReg[i] = *(const float4*)(bbase + (size_t)(k0 + kh + i) * HID);
        }

        // ---- compute from current buffer ----
        v16bf afr[2];
#pragma unroll
        for (int im = 0; im < 2; ++im) {
            const __bf16* p = &As[cur][(amRow + im * 16) * LDA + khA];
            v8bf lo = *(const v8bf*)p;          // K: khA+0..7
            v8bf hi = *(const v8bf*)(p + 16);   // K: 16+khA+0..7
            v16bf t;
#pragma unroll
            for (int q = 0; q < 8; ++q) { t[q] = lo[q]; t[q + 8] = hi[q]; }
            afr[im] = t;
        }
#pragma unroll
        for (int in = 0; in < 4; ++in) {
            const __bf16* p = &Bs[cur][(bnRow + in * 16) * LDA + khB];
            v8bf lo = *(const v8bf*)p;          // K: khB+0..7
            v8bf hi = *(const v8bf*)(p + 16);   // K: 16+khB+0..7
            v16bf bf;
#pragma unroll
            for (int q = 0; q < 8; ++q) { bf[q] = lo[q]; bf[q + 8] = hi[q]; }
#pragma unroll
            for (int im = 0; im < 2; ++im)
                acc[im][in] = __builtin_amdgcn_wmma_f32_16x16x32_bf16(
                    false, afr[im], false, bf, (short)0, acc[im][in], false, false);
        }

        // ---- stage prefetched slice into the other buffer ----
        if (kt + 1 < KT) {
#pragma unroll
            for (int p = 0; p < 4; ++p) {
                v4bf h = { (__bf16)aReg[p].x, (__bf16)aReg[p].y, (__bf16)aReg[p].z, (__bf16)aReg[p].w };
                *(v4bf*)&As[nxt][(rA + 32 * p) * LDA + cA] = h;
            }
            v4bf h0 = { (__bf16)bReg[0].x, (__bf16)bReg[1].x, (__bf16)bReg[2].x, (__bf16)bReg[3].x };
            v4bf h1 = { (__bf16)bReg[0].y, (__bf16)bReg[1].y, (__bf16)bReg[2].y, (__bf16)bReg[3].y };
            v4bf h2 = { (__bf16)bReg[0].z, (__bf16)bReg[1].z, (__bf16)bReg[2].z, (__bf16)bReg[3].z };
            v4bf h3 = { (__bf16)bReg[0].w, (__bf16)bReg[1].w, (__bf16)bReg[2].w, (__bf16)bReg[3].w };
            *(v4bf*)&Bs[nxt][(nq + 0) * LDA + kh] = h0;
            *(v4bf*)&Bs[nxt][(nq + 1) * LDA + kh] = h1;
            *(v4bf*)&Bs[nxt][(nq + 2) * LDA + kh] = h2;
            *(v4bf*)&Bs[nxt][(nq + 3) * LDA + kh] = h3;
        }
        __syncthreads();
    }

    // ---- epilogue: scale by gate weight, add bias, scatter (2 commutative
    //      atomic adds per output element -> bitwise deterministic) ----
#pragma unroll
    for (int in = 0; in < 4; ++in) {
        int col = n0 + wn + in * 16 + (lane & 15);
        float bv = bias[e * HID + col];
#pragma unroll
        for (int im = 0; im < 2; ++im) {
            int mb = wm + im * 16 + (lane >> 4) * 8;   // C layout: lanes16-31 -> M+8
#pragma unroll
            for (int r = 0; r < 8; ++r) {
                int m = mb + r;
                float w = wgtL[m];
                if (w != 0.f) {
                    float val = w * (acc[im][in][r] + bv);
                    atomicAdd(&y[(size_t)tokL[m] * HID + col], val);
                }
            }
        }
    }
}

// ---------------------------------------------------------------------------
// Host launcher
// ---------------------------------------------------------------------------
extern "C" void kernel_launch(void* const* d_in, const int* in_sizes, int n_in,
                              void* d_out, int out_size, void* d_ws, size_t ws_size,
                              hipStream_t stream) {
    const float* x  = (const float*)d_in[0];
    const float* Wg = (const float*)d_in[1];
    const float* bg = (const float*)d_in[2];
    const float* W  = (const float*)d_in[3];
    const float* b  = (const float*)d_in[4];
    float* y = (float*)d_out;

    // workspace layout: counts(64 ints, padded) | tok[E*N] | wgt[E*N]  (~512KB)
    int*   counts = (int*)d_ws;
    int*   tok    = counts + 64;
    float* wgt    = (float*)(tok + N_EXP * N_TOK);

    // 1) zero output + counters
    moe_zero_kernel<<<(N_TOK * HID / 4 + 255) / 256, 256, 0, stream>>>(y, counts);
    // 2) gating / routing: one wave per token
    moe_gating_kernel<<<N_TOK / 8, 256, 0, stream>>>(x, Wg, bg, counts, tok, wgt);
    // 3) routed expert GEMMs (bf16 WMMA)
    moe_gemm_kernel<<<dim3(HID / BN, N_TOK / BM, N_EXP), 256, 0, stream>>>(
        x, W, b, counts, tok, wgt, y);
}
